// CausalSelfAttention_64252710748295
// MI455X (gfx1250) — matrix operand order repaired and checked by reading, not compiled, not run on hardware
//
#include <hip/hip_runtime.h>

#define D_MODEL 512
#define SEQ     4096
#define BATCH   2
#define NHEAD   8
#define HD      64

typedef __attribute__((ext_vector_type(16))) __bf16 v16bf;
typedef __attribute__((ext_vector_type(8)))  __bf16 v8bf;
typedef __attribute__((ext_vector_type(8)))  float  v8f;

// D = A(16x32 bf16) * B(32x16 bf16) + C(16x16 f32)
static __device__ __forceinline__ v8f wmma_bf16(v16bf a, v16bf b, v8f c) {
  return __builtin_amdgcn_wmma_f32_16x16x32_bf16(false, a, false, b, (short)0, c,
                                                 false, false);
}

// A-matrix 16x32 bf16 fragment from row-major [rows][ld] storage.
// ISA layout: lane L (m=L&15, h=L>>4) holds row m, elements e<8 -> K=8h+e,
// e>=8 -> K=16+8h+(e-8): two contiguous 16-byte runs.
static __device__ __forceinline__ v16bf load_A_bf16(const __bf16* base, int row0,
                                                    int k0, int ld) {
  const int lane = threadIdx.x & 31;
  const int m = lane & 15, h = lane >> 4;
  const __bf16* p = base + (size_t)(row0 + m) * ld + (k0 + 8 * h);
  v8bf lo = *(const v8bf*)(p);
  v8bf hi = *(const v8bf*)(p + 16);
  v16bf r;
#pragma unroll
  for (int i = 0; i < 8; ++i) { r[i] = lo[i]; r[i + 8] = hi[i]; }
  return r;
}

// B-matrix 32x16 bf16 fragment; B element (k, n) read from TRANSPOSED storage
// baseT[n][k] (row stride ld). ISA layout: lane L (n=L&15, h=L>>4) holds col n,
// e<8 -> K=k0+8h+e, e>=8 -> K=k0+16+8h+(e-8): two contiguous 16-byte runs.
static __device__ __forceinline__ v16bf load_B_bf16(const __bf16* baseT, int n0,
                                                    int k0, int ld) {
  const int lane = threadIdx.x & 31;
  const int n = lane & 15, h = lane >> 4;
  const __bf16* p = baseT + (size_t)(n0 + n) * ld + (k0 + 8 * h);
  v8bf lo = *(const v8bf*)(p);
  v8bf hi = *(const v8bf*)(p + 16);
  v16bf r;
#pragma unroll
  for (int i = 0; i < 8; ++i) { r[i] = lo[i]; r[i + 8] = hi[i]; }
  return r;
}

// ---------------- pack kernels ----------------
__global__ void pack_x_kernel(const float* __restrict__ x, __bf16* __restrict__ xb,
                              int n) {
  int i = blockIdx.x * 256 + threadIdx.x;
  if (i < n) xb[i] = (__bf16)x[i];
}

// Wt[n][k] = (bf16) W[k][n], D_MODEL x D_MODEL
__global__ void pack_wt_kernel(const float* __restrict__ W, __bf16* __restrict__ Wt) {
  int k = blockIdx.x * 32 + (threadIdx.x & 31);
  int n = blockIdx.y * 8 + (threadIdx.x >> 5);
  Wt[(size_t)n * D_MODEL + k] = (__bf16)W[(size_t)k * D_MODEL + n];
}

// ---------------- QKV projection ----------------
// One 16(m) x 64(n) output strip per wave: 4 accumulators share one A fragment
// per k-step (4 WMMAs per A load). 4 waves per block along n.
// which = blockIdx.z : 0->Q, 1->K (row-major [b,h,s,d]), 2->V transposed [b,h,d,s]
__global__ void gemm_qkv_kernel(const __bf16* __restrict__ xb,
                                const __bf16* __restrict__ Wqt,
                                const __bf16* __restrict__ Wkt,
                                const __bf16* __restrict__ Wvt,
                                const float* __restrict__ bq,
                                const float* __restrict__ bk,
                                const float* __restrict__ bv,
                                __bf16* __restrict__ Qo, __bf16* __restrict__ Ko,
                                __bf16* __restrict__ Vto) {
  const int wave = threadIdx.x >> 5;
  const int lane = threadIdx.x & 31;
  const int nl = lane & 15, half = lane >> 4;
  const int n0 = (blockIdx.x * 4 + wave) * 64;
  const int m0 = blockIdx.y * 16;
  const int which = blockIdx.z;
  const __bf16* Wt = (which == 0) ? Wqt : (which == 1) ? Wkt : Wvt;
  const float* bias = (which == 0) ? bq : (which == 1) ? bk : bv;

  v8f c[4];
#pragma unroll
  for (int j = 0; j < 4; ++j) {
    const float bn = bias[n0 + j * 16 + nl];   // C layout: lane holds fixed n
#pragma unroll
    for (int r = 0; r < 8; ++r) c[j][r] = bn;
  }

#pragma unroll 2
  for (int k0 = 0; k0 < D_MODEL; k0 += 32) {
    const v16bf a = load_A_bf16(xb, m0, k0, D_MODEL);
#pragma unroll
    for (int j = 0; j < 4; ++j)
      c[j] = wmma_bf16(a, load_B_bf16(Wt, n0 + j * 16, k0, D_MODEL), c[j]);
  }

#pragma unroll
  for (int j = 0; j < 4; ++j) {
    const int ng = n0 + j * 16 + nl, h = ng >> 6, d = ng & (HD - 1);
#pragma unroll
    for (int r = 0; r < 8; ++r) {
      const int mg = m0 + r + 8 * half;            // row in [B*S]
      const int b_ = mg >> 12, s = mg & (SEQ - 1);
      const __bf16 v = (__bf16)c[j][r];
      if (which == 0)
        Qo[((size_t)(b_ * NHEAD + h) * SEQ + s) * HD + d] = v;
      else if (which == 1)
        Ko[((size_t)(b_ * NHEAD + h) * SEQ + s) * HD + d] = v;
      else
        Vto[((size_t)(b_ * NHEAD + h) * HD + d) * SEQ + s] = v;
    }
  }
}

// ---------------- flash attention ----------------
// One 16-query tile per wave (4 waves/block), online softmax over 32-key chunks.
__global__ void attn_kernel(const __bf16* __restrict__ Q, const __bf16* __restrict__ K,
                            const __bf16* __restrict__ Vt, __bf16* __restrict__ AO) {
  const int wave = threadIdx.x >> 5;
  const int lane = threadIdx.x & 31;
  const int nl = lane & 15, half = lane >> 4;
  const int q0 = (blockIdx.x * 4 + wave) * 16;
  const int bh = blockIdx.y;
  const int b = bh >> 3, h = bh & 7;

  const __bf16* Qh = Q + (size_t)bh * SEQ * HD;   // [s][d]
  const __bf16* Kh = K + (size_t)bh * SEQ * HD;   // [s][d]  (B frag for Q@K^T)
  const __bf16* Vh = Vt + (size_t)bh * HD * SEQ;  // [d][s]  (B frag for P@V)

  __shared__ __align__(64) __bf16 Pl[4][16 * 32];
  __bf16* Pw = &Pl[wave][0];

  const v16bf aq0 = load_A_bf16(Qh, q0, 0, HD);
  const v16bf aq1 = load_A_bf16(Qh, q0, 32, HD);

  v8f o0 = {}, o1 = {}, o2 = {}, o3 = {};
  float rmax[8], rsum[8];
#pragma unroll
  for (int r = 0; r < 8; ++r) { rmax[r] = -3.0e38f; rsum[r] = 0.f; }

  const int kend = q0 + 16;          // causal: keys < kend
  const float scale = 0.125f;        // hd^-0.5

  for (int k0 = 0; k0 < kend; k0 += 32) {
    if (k0 + 32 < kend) {
      __builtin_prefetch(Kh + (size_t)(k0 + 32) * HD, 0, 1);
      __builtin_prefetch(Vh + (size_t)(k0 + 32), 0, 1);
    }
    const bool t1valid = (k0 + 16) < kend;      // wave-uniform, EXEC stays full

    v8f s0 = {}, s1 = {};
    s0 = wmma_bf16(aq0, load_B_bf16(Kh, k0, 0, HD), s0);
    s0 = wmma_bf16(aq1, load_B_bf16(Kh, k0, 32, HD), s0);
    if (t1valid) {
      s1 = wmma_bf16(aq0, load_B_bf16(Kh, k0 + 16, 0, HD), s1);
      s1 = wmma_bf16(aq1, load_B_bf16(Kh, k0 + 16, 32, HD), s1);
    }

    const bool need_mask = (k0 + 31) > q0;
#pragma unroll
    for (int r = 0; r < 8; ++r) {
      const int m = r + 8 * half;               // score row (query)
      const int q = q0 + m;
      float v0 = s0[r] * scale;
      float v1 = t1valid ? s1[r] * scale : -3.0e38f;
      if (need_mask) {
        if (k0 + nl > q) v0 = -3.0e38f;
        if (k0 + 16 + nl > q) v1 = -3.0e38f;
      }
      float cm = fmaxf(v0, v1);
#pragma unroll
      for (int off = 8; off >= 1; off >>= 1) cm = fmaxf(cm, __shfl_xor(cm, off, 16));
      const float nm = fmaxf(rmax[r], cm);
      const float al = __expf(rmax[r] - nm);
      rmax[r] = nm;
      const float p0 = __expf(v0 - nm);
      const float p1 = __expf(v1 - nm);
      float ps = p0 + p1;
#pragma unroll
      for (int off = 8; off >= 1; off >>= 1) ps += __shfl_xor(ps, off, 16);
      rsum[r] = rsum[r] * al + ps;
      o0[r] *= al; o1[r] *= al; o2[r] *= al; o3[r] *= al;
      // stage P (C layout -> row-major LDS) for A-fragment reload
      Pw[m * 32 + nl] = (__bf16)p0;
      Pw[m * 32 + 16 + nl] = (__bf16)p1;
    }
    asm volatile("s_wait_dscnt 0x0" ::: "memory");  // wave-local LDS RAW fence

    const v16bf ap = load_A_bf16(Pw, 0, 0, 32);     // P tile 16x32
    o0 = wmma_bf16(ap, load_B_bf16(Vh, 0, k0, SEQ), o0);
    o1 = wmma_bf16(ap, load_B_bf16(Vh, 16, k0, SEQ), o1);
    o2 = wmma_bf16(ap, load_B_bf16(Vh, 32, k0, SEQ), o2);
    o3 = wmma_bf16(ap, load_B_bf16(Vh, 48, k0, SEQ), o3);
    asm volatile("s_wait_dscnt 0x0" ::: "memory");  // drain before Pw overwrite
  }

#pragma unroll
  for (int r = 0; r < 8; ++r) {
    const int m = r + 8 * half;
    const float inv = 1.0f / rsum[r];
    __bf16* dst = AO + (size_t)(b * SEQ + q0 + m) * D_MODEL + h * HD;
    dst[0 + nl]  = (__bf16)(o0[r] * inv);
    dst[16 + nl] = (__bf16)(o1[r] * inv);
    dst[32 + nl] = (__bf16)(o2[r] * inv);
    dst[48 + nl] = (__bf16)(o3[r] * inv);
  }
}

// ---------------- output projection (f32 out) ----------------
// Same 16x64-per-wave register blocking as the QKV GEMM.
__global__ void gemm_out_kernel(const __bf16* __restrict__ AO,
                                const __bf16* __restrict__ Wot,
                                const float* __restrict__ bo,
                                float* __restrict__ out) {
  const int wave = threadIdx.x >> 5;
  const int lane = threadIdx.x & 31;
  const int nl = lane & 15, half = lane >> 4;
  const int n0 = (blockIdx.x * 4 + wave) * 64;
  const int m0 = blockIdx.y * 16;

  v8f c[4];
#pragma unroll
  for (int j = 0; j < 4; ++j) {
    const float bn = bo[n0 + j * 16 + nl];
#pragma unroll
    for (int r = 0; r < 8; ++r) c[j][r] = bn;
  }

#pragma unroll 2
  for (int k0 = 0; k0 < D_MODEL; k0 += 32) {
    const v16bf a = load_A_bf16(AO, m0, k0, D_MODEL);
#pragma unroll
    for (int j = 0; j < 4; ++j)
      c[j] = wmma_bf16(a, load_B_bf16(Wot, n0 + j * 16, k0, D_MODEL), c[j]);
  }

#pragma unroll
  for (int j = 0; j < 4; ++j) {
#pragma unroll
    for (int r = 0; r < 8; ++r) {
      const int mg = m0 + r + 8 * half;
      out[(size_t)mg * D_MODEL + (n0 + j * 16 + nl)] = c[j][r];
    }
  }
}

extern "C" void kernel_launch(void* const* d_in, const int* in_sizes, int n_in,
                              void* d_out, int out_size, void* d_ws, size_t ws_size,
                              hipStream_t stream) {
  (void)in_sizes; (void)n_in; (void)out_size; (void)ws_size;
  const float* x  = (const float*)d_in[0];
  const float* Wq = (const float*)d_in[1];
  const float* bq = (const float*)d_in[2];
  const float* Wk = (const float*)d_in[3];
  const float* bk = (const float*)d_in[4];
  const float* Wv = (const float*)d_in[5];
  const float* bv = (const float*)d_in[6];
  const float* Wo = (const float*)d_in[7];
  const float* bo = (const float*)d_in[8];
  float* out = (float*)d_out;

  char* ws = (char*)d_ws;
  size_t off = 0;
  auto alloc = [&](size_t bytes) -> void* {
    void* p = ws + off;
    off = (off + bytes + 255) & ~(size_t)255;
    return p;
  };
  const size_t actBytes = (size_t)BATCH * SEQ * D_MODEL * sizeof(__bf16);  // 8 MB
  const size_t wBytes   = (size_t)D_MODEL * D_MODEL * sizeof(__bf16);
  __bf16* xb  = (__bf16*)alloc(actBytes);
  __bf16* Wqt = (__bf16*)alloc(wBytes);
  __bf16* Wkt = (__bf16*)alloc(wBytes);
  __bf16* Wvt = (__bf16*)alloc(wBytes);
  __bf16* Wot = (__bf16*)alloc(wBytes);
  __bf16* Qb  = (__bf16*)alloc(actBytes);
  __bf16* Kb  = (__bf16*)alloc(actBytes);
  __bf16* Vtb = (__bf16*)alloc(actBytes);
  __bf16* AOb = (__bf16*)alloc(actBytes);   // total ~42 MB of ws

  const int nElems = BATCH * SEQ * D_MODEL;
  pack_x_kernel<<<dim3((nElems + 255) / 256), dim3(256), 0, stream>>>(x, xb, nElems);
  dim3 wtg(D_MODEL / 32, D_MODEL / 8);
  pack_wt_kernel<<<wtg, dim3(256), 0, stream>>>(Wq, Wqt);
  pack_wt_kernel<<<wtg, dim3(256), 0, stream>>>(Wk, Wkt);
  pack_wt_kernel<<<wtg, dim3(256), 0, stream>>>(Wv, Wvt);
  pack_wt_kernel<<<wtg, dim3(256), 0, stream>>>(Wo, Wot);

  // QKV: each wave does a 16x64 strip; block covers 256 cols -> grid.x = 2
  gemm_qkv_kernel<<<dim3(D_MODEL / 256, (BATCH * SEQ) / 16, 3), dim3(128), 0, stream>>>(
      xb, Wqt, Wkt, Wvt, bq, bk, bv, Qb, Kb, Vtb);

  // attention: q-tiles 256 (4/block), bh = 16
  attn_kernel<<<dim3(SEQ / 64, BATCH * NHEAD), dim3(128), 0, stream>>>(Qb, Kb, Vtb, AOb);

  // output projection
  gemm_out_kernel<<<dim3(D_MODEL / 256, (BATCH * SEQ) / 16), dim3(128), 0, stream>>>(
      AOb, Wot, bo, out);
}
